// Decoder_7481833029763
// MI455X (gfx1250) — matrix operand order repaired and checked by reading, not compiled, not run on hardware
//
#include <hip/hip_runtime.h>
#include <hip/hip_bf16.h>
#include <cstdint>
#include <cstddef>

typedef __attribute__((ext_vector_type(16))) _Float16 v16h;
typedef __attribute__((ext_vector_type(8)))  _Float16 v8h;
typedef __attribute__((ext_vector_type(8)))  float    v8f;

namespace {
constexpr int T_STEPS = 1000;   // encoder time steps
constexpr int NB      = 64;     // batch
constexpr int LSEQ    = 250;    // decoder steps
constexpr int HDIM    = 512;    // LSTM1 hidden
constexpr int KDIM    = 128;    // LSTM2 hidden / key dim
constexpr int VDIM    = 128;    // value dim
constexpr int NVOC    = 34;     // vocab
constexpr int IN1     = HDIM + VDIM + HDIM; // 1152: [emb | ctx | h1]
constexpr int IN2     = HDIM + KDIM;        // 640:  [h1 | h2]
}

__device__ __forceinline__ float sigf(float x) { return 1.0f / (1.0f + expf(-x)); }

// ---------------------------------------------------------------------------
// Fused gate-GEMM + LSTM cell.
//   gates = X(64 x IN) @ W(4*GU x IN)^T + bias;  LSTM nonlinearity; update c,h.
// One block owns 16 hidden units (h0..h0+15) across ALL 4 gates and all 64
// batch rows: 4 m-tiles x 4 gate-tiles = 16 WMMA output tiles, K = IN.
// 8 waves: wave w -> m-tile (w>>1), gate pair (w&1)*2 + {0,1}.
// ---------------------------------------------------------------------------
__global__ __launch_bounds__(256) void lstm_gemm_cell(
    const _Float16* __restrict__ X,     // (64, IN)
    const _Float16* __restrict__ W,     // (4*GU, IN) row-major
    const float*    __restrict__ bias,  // (4*GU)  b_ih + b_hh
    float*          __restrict__ cstate,// (64, GU)
    _Float16*       __restrict__ hA, int ldA, int offA,   // f16 h dest (required)
    _Float16*       __restrict__ hB, int ldB, int offB,   // optional 2nd f16 dest
    float*          __restrict__ hf32,                    // optional f32 h dest
    int GU, int IN)
{
  __shared__ float lds_g[4 * 64 * 16];   // [gate][n][hh]  16 KB

  const int tid   = threadIdx.x;
  const int w     = tid >> 5;
  const int lane  = tid & 31;
  const int h0    = blockIdx.x * 16;
  const int mt    = w >> 1;              // m-tile 0..3
  const int gp    = w & 1;               // gate pair
  const int lhalf = lane >> 4;           // 0: lanes 0-15, 1: lanes 16-31
  const int ncol  = lane & 15;
  const int koffA = lhalf * 16;          // A-frag K offset, contiguous order k = 16h + s
  const int kloB  = lhalf * 16;          // B-frag K offset, contiguous order k = 16h + s
  const int mrow  = mt * 16 + ncol;      // A row for this lane
  const int g0    = gp * 2, g1 = gp * 2 + 1;

  const _Float16* xrow  = X + (size_t)mrow * IN;
  const _Float16* w0row = W + (size_t)(g0 * GU + h0 + ncol) * IN;
  const _Float16* w1row = W + (size_t)(g1 * GU + h0 + ncol) * IN;

  v8f acc0 = {};
  v8f acc1 = {};

  for (int ks = 0; ks < IN; ks += 32) {
    // A fragment: lane holds M=mrow, K = ks + koffA + 0..15 (contiguous)
    v8h alo = *(const v8h*)(xrow + ks + koffA);
    v8h ahi = *(const v8h*)(xrow + ks + koffA + 8);
    v16h a;
#pragma unroll
    for (int i = 0; i < 8; ++i) { a[i] = alo[i]; a[8 + i] = ahi[i]; }

    // B fragments: lane holds col N=ncol, rows K = ks + kloB + 0..15 (contiguous in W row)
    v8h b0l = *(const v8h*)(w0row + ks + kloB);
    v8h b0h = *(const v8h*)(w0row + ks + kloB + 8);
    v8h b1l = *(const v8h*)(w1row + ks + kloB);
    v8h b1h = *(const v8h*)(w1row + ks + kloB + 8);
    v16h b0, b1;
#pragma unroll
    for (int i = 0; i < 8; ++i) {
      b0[i] = b0l[i]; b0[8 + i] = b0h[i];
      b1[i] = b1l[i]; b1[8 + i] = b1h[i];
    }

    acc0 = __builtin_amdgcn_wmma_f32_16x16x32_f16(false, a, false, b0,
                                                  (short)0, acc0, false, false);
    acc1 = __builtin_amdgcn_wmma_f32_16x16x32_f16(false, a, false, b1,
                                                  (short)0, acc1, false, false);
  }

  // Stage accumulators: D layout -> VGPR r, lanes<16: M=r, lanes>=16: M=8+r; N=ncol
#pragma unroll
  for (int r = 0; r < 8; ++r) {
    const int n = mt * 16 + r + (lhalf ? 8 : 0);
    lds_g[((g0 * 64) + n) * 16 + ncol] = acc0[r];
    lds_g[((g1 * 64) + n) * 16 + ncol] = acc1[r];
  }
  __syncthreads();

  // Elementwise LSTM cell over 64 x 16 outputs
  for (int idx = tid; idx < 64 * 16; idx += 256) {
    const int n  = idx >> 4;
    const int hh = idx & 15;
    const int h  = h0 + hh;
    const float gi = lds_g[((0 * 64) + n) * 16 + hh] + bias[0 * GU + h];
    const float gf = lds_g[((1 * 64) + n) * 16 + hh] + bias[1 * GU + h];
    const float gg = lds_g[((2 * 64) + n) * 16 + hh] + bias[2 * GU + h];
    const float go = lds_g[((3 * 64) + n) * 16 + hh] + bias[3 * GU + h];
    const float cprev = cstate[(size_t)n * GU + h];
    const float cn = sigf(gf) * cprev + sigf(gi) * tanhf(gg);
    const float hn = sigf(go) * tanhf(cn);
    cstate[(size_t)n * GU + h] = cn;
    hA[(size_t)n * ldA + offA + h] = (_Float16)hn;
    if (hB)   hB[(size_t)n * ldB + offB + h] = (_Float16)hn;
    if (hf32) hf32[(size_t)n * GU + h] = hn;
  }
}

// ---------------------------------------------------------------------------
// Attention + softmax + context + output projection + pack next-step input.
// One block per batch row n. L2-resident key/value streaming, softmax in LDS.
// ---------------------------------------------------------------------------
__global__ __launch_bounds__(256) void attn_step(
    const float* __restrict__ key_enc,  // (T, N, K)
    const float* __restrict__ values,   // (T, N, V)
    const int*   __restrict__ lens,     // (N)
    const int*   __restrict__ text,     // (N, L)
    const float* __restrict__ emb,      // (VOCAB, H)
    const float* __restrict__ Wout,     // (34, K+V)
    const float* __restrict__ bout,     // (34)
    const float* __restrict__ h2f32,    // (N, K)
    _Float16*    __restrict__ xh1_next, // (N, IN1) next-step LSTM1 input
    float*       __restrict__ preds,    // (N, L, 34)
    float*       __restrict__ atts,     // (N, L, T)
    int tstep)
{
  __shared__ float sh_e[T_STEPS];
  __shared__ float sh_h2[KDIM];
  __shared__ float sh_red[256];
  __shared__ float sh_ctx[256];

  const int n   = blockIdx.x;
  const int tid = threadIdx.x;

  for (int j = tid; j < KDIM; j += 256) sh_h2[j] = h2f32[n * KDIM + j];
  __syncthreads();

  const int len = lens[n];

  // energy + row max
  float lmax = -3.0e38f;
  for (int t = tid; t < T_STEPS; t += 256) {
    float e;
    if (t >= len) {
      e = -1.0e9f;
    } else {
      const float* kp = key_enc + ((size_t)t * NB + n) * KDIM;
      float s = 0.f;
#pragma unroll 4
      for (int k = 0; k < KDIM; ++k) s += kp[k] * sh_h2[k];
      e = s;
    }
    sh_e[t] = e;
    lmax = fmaxf(lmax, e);
  }
  sh_red[tid] = lmax; __syncthreads();
  for (int s = 128; s > 0; s >>= 1) {
    if (tid < s) sh_red[tid] = fmaxf(sh_red[tid], sh_red[tid + s]);
    __syncthreads();
  }
  const float mx = sh_red[0];
  __syncthreads();

  // exp + sum
  float lsum = 0.f;
  for (int t = tid; t < T_STEPS; t += 256) {
    const float v = expf(sh_e[t] - mx);
    sh_e[t] = v;
    lsum += v;
  }
  sh_red[tid] = lsum; __syncthreads();
  for (int s = 128; s > 0; s >>= 1) {
    if (tid < s) sh_red[tid] += sh_red[tid + s];
    __syncthreads();
  }
  const float inv = 1.0f / sh_red[0];
  __syncthreads();

  // normalize + write attentions
  float* attrow = atts + ((size_t)n * LSEQ + tstep) * T_STEPS;
  for (int t = tid; t < T_STEPS; t += 256) {
    const float a = sh_e[t] * inv;
    sh_e[t] = a;
    attrow[t] = a;
  }
  __syncthreads();

  // context: ctx[v] = sum_t att[t] * values[t][n][v]  (2 partial sums per v)
  const int v    = tid & (VDIM - 1);
  const int half = tid >> 7;   // 0 or 1
  float cacc = 0.f;
  for (int t = half; t < T_STEPS; t += 2)
    cacc += sh_e[t] * values[((size_t)t * NB + n) * VDIM + v];
  sh_ctx[tid] = cacc; __syncthreads();
  if (tid < VDIM) sh_ctx[tid] += sh_ctx[tid + VDIM];
  __syncthreads();

  // output projection: pred = [h2 | ctx] @ Wout^T + bout
  if (tid < NVOC) {
    float p = bout[tid];
    const float* wrow = Wout + tid * (KDIM + VDIM);
    for (int j = 0; j < KDIM; ++j) p += sh_h2[j] * wrow[j];
    for (int j = 0; j < VDIM; ++j) p += sh_ctx[j] * wrow[KDIM + j];
    preds[((size_t)n * LSEQ + tstep) * NVOC + tid] = p;
  }

  // pack next step input: [emb_{t+1} | ctx_t | (h1 filled by next gemm1's producer)]
  for (int j = tid; j < VDIM; j += 256)
    xh1_next[(size_t)n * IN1 + HDIM + j] = (_Float16)sh_ctx[j];
  if (tstep + 1 < LSEQ) {
    const int tok = text[n * LSEQ + tstep + 1];
    const float* erow = emb + (size_t)tok * HDIM;
    for (int j = tid; j < HDIM; j += 256)
      xh1_next[(size_t)n * IN1 + j] = (_Float16)erow[j];
  }
}

// ---------------------------------------------------------------------------
// One-time setup kernels
// ---------------------------------------------------------------------------
__global__ void init_wb(const float* __restrict__ Wih1, const float* __restrict__ Whh1,
                        const float* __restrict__ bih1, const float* __restrict__ bhh1,
                        const float* __restrict__ Wih2, const float* __restrict__ Whh2,
                        const float* __restrict__ bih2, const float* __restrict__ bhh2,
                        _Float16* __restrict__ W1, _Float16* __restrict__ W2,
                        float* __restrict__ b1, float* __restrict__ b2)
{
  const int stride = gridDim.x * blockDim.x;
  const int tid0   = blockIdx.x * blockDim.x + threadIdx.x;
  const int tot1 = 4 * HDIM * IN1;
  for (int i = tid0; i < tot1; i += stride) {
    const int j = i / IN1, k = i - j * IN1;
    const float x = (k < HDIM + VDIM) ? Wih1[j * (HDIM + VDIM) + k]
                                      : Whh1[j * HDIM + (k - (HDIM + VDIM))];
    W1[i] = (_Float16)x;
  }
  const int tot2 = 4 * KDIM * IN2;
  for (int i = tid0; i < tot2; i += stride) {
    const int j = i / IN2, k = i - j * IN2;
    const float x = (k < HDIM) ? Wih2[j * HDIM + k]
                               : Whh2[j * KDIM + (k - HDIM)];
    W2[i] = (_Float16)x;
  }
  for (int i = tid0; i < 4 * HDIM; i += stride) b1[i] = bih1[i] + bhh1[i];
  for (int i = tid0; i < 4 * KDIM; i += stride) b2[i] = bih2[i] + bhh2[i];
}

__global__ void zero_mem(uint32_t* __restrict__ p, size_t nwords)
{
  const size_t stride = (size_t)gridDim.x * blockDim.x;
  for (size_t i = (size_t)blockIdx.x * blockDim.x + threadIdx.x; i < nwords; i += stride)
    p[i] = 0u;
}

__global__ void init_x0(const int* __restrict__ text, const float* __restrict__ emb,
                        const float* __restrict__ values, _Float16* __restrict__ xh1)
{
  const int n = blockIdx.x, tid = threadIdx.x;
  const int tok = text[n * LSEQ + 0];
  for (int j = tid; j < HDIM; j += 256)
    xh1[(size_t)n * IN1 + j] = (_Float16)emb[(size_t)tok * HDIM + j];
  for (int j = tid; j < VDIM; j += 256)
    xh1[(size_t)n * IN1 + HDIM + j] = (_Float16)values[((size_t)0 * NB + n) * VDIM + j];
  for (int j = tid; j < HDIM; j += 256)
    xh1[(size_t)n * IN1 + HDIM + VDIM + j] = (_Float16)0.0f;   // h1_{-1} = 0
}

// ---------------------------------------------------------------------------
extern "C" void kernel_launch(void* const* d_in, const int* in_sizes, int n_in,
                              void* d_out, int out_size, void* d_ws, size_t ws_size,
                              hipStream_t stream)
{
  (void)in_sizes; (void)n_in; (void)out_size; (void)ws_size;

  const float* key_enc = (const float*)d_in[0];
  const float* values  = (const float*)d_in[1];
  const int*   lens    = (const int*)d_in[2];
  const int*   text    = (const int*)d_in[3];
  const float* emb     = (const float*)d_in[4];
  const float* Wih1    = (const float*)d_in[5];
  const float* Whh1    = (const float*)d_in[6];
  const float* bih1    = (const float*)d_in[7];
  const float* bhh1    = (const float*)d_in[8];
  const float* Wih2    = (const float*)d_in[9];
  const float* Whh2    = (const float*)d_in[10];
  const float* bih2    = (const float*)d_in[11];
  const float* bhh2    = (const float*)d_in[12];
  const float* Wout    = (const float*)d_in[13];
  const float* bout    = (const float*)d_in[14];

  float* preds = (float*)d_out;
  float* atts  = preds + (size_t)NB * LSEQ * NVOC;

  // workspace carve-out
  size_t off = 0;
  auto take = [&](size_t bytes) -> void* {
    void* p = (char*)d_ws + off;
    off += (bytes + 255) & ~(size_t)255;
    return p;
  };
  _Float16* W1 = (_Float16*)take((size_t)4 * HDIM * IN1 * sizeof(_Float16)); // 4.7 MB
  _Float16* W2 = (_Float16*)take((size_t)4 * KDIM * IN2 * sizeof(_Float16)); // 0.66 MB
  float* b1 = (float*)take((size_t)4 * HDIM * sizeof(float));
  float* b2 = (float*)take((size_t)4 * KDIM * sizeof(float));

  const size_t stateOff = off;
  _Float16* xh1[2] = { (_Float16*)take((size_t)NB * IN1 * sizeof(_Float16)),
                       (_Float16*)take((size_t)NB * IN1 * sizeof(_Float16)) };
  _Float16* xh2[2] = { (_Float16*)take((size_t)NB * IN2 * sizeof(_Float16)),
                       (_Float16*)take((size_t)NB * IN2 * sizeof(_Float16)) };
  float* c1    = (float*)take((size_t)NB * HDIM * sizeof(float));
  float* c2    = (float*)take((size_t)NB * KDIM * sizeof(float));
  float* h2f32 = (float*)take((size_t)NB * KDIM * sizeof(float));
  const size_t stateBytes = off - stateOff;

  // one-time init
  init_wb<<<512, 256, 0, stream>>>(Wih1, Whh1, bih1, bhh1, Wih2, Whh2, bih2, bhh2,
                                   W1, W2, b1, b2);
  zero_mem<<<256, 256, 0, stream>>>((uint32_t*)((char*)d_ws + stateOff), stateBytes / 4);
  init_x0<<<NB, 256, 0, stream>>>(text, emb, values, xh1[0]);

  // sequential decode, double-buffered activations by step parity
  for (int t = 0; t < LSEQ; ++t) {
    const int p = t & 1, q = p ^ 1;
    // LSTM1: reads xh1[p] = [emb_t | ctx_{t-1} | h1_{t-1}]
    //        writes h1_t -> xh1[q] (cols 640..) and xh2[p] (cols 0..511)
    lstm_gemm_cell<<<HDIM / 16, 256, 0, stream>>>(
        xh1[p], W1, b1, c1,
        xh1[q], IN1, HDIM + VDIM,
        xh2[p], IN2, 0,
        nullptr, HDIM, IN1);
    // LSTM2: reads xh2[p] = [h1_t | h2_{t-1}]
    //        writes h2_t -> xh2[q] (cols 512..) and h2f32
    lstm_gemm_cell<<<KDIM / 16, 256, 0, stream>>>(
        xh2[p], W2, b2, c2,
        xh2[q], IN2, HDIM,
        nullptr, 0, 0,
        h2f32, KDIM, IN2);
    // attention + softmax + ctx + pred; packs emb_{t+1}, ctx_t into xh1[q]
    attn_step<<<NB, 256, 0, stream>>>(
        key_enc, values, lens, text, emb, Wout, bout,
        h2f32, xh1[q], preds, atts, t);
  }
}
